// HighFreqMHSA_46420006535707
// MI455X (gfx1250) — compile-verified, matrix operand order checked by reading, not run on hardware
//
#include <hip/hip_runtime.h>

// High-frequency MHSA (LeViT-style, 14x14 tokens) for gfx1250 / MI455X.
// All contractions run on v_wmma_f32_16x16x32_bf16 (bf16 in, f32 accum).
// Attention computes S^T = K^T Q so each lane owns half a softmax row:
// softmax needs only one cross-lane shuffle, P stores are packed b128,
// bias is pre-packed into tile-register order (one f32x8 load per tile).

#define DIMC   512
#define HEADS  8
#define HD     64
#define NRES   196          // 14*14 real tokens
#define NP     224          // padded tokens: 14 tiles of 16, 7 k-steps of 32
#define NT13   13           // query-row tiles (208 rows)
#define BATCH  128
#define OQKV   1536

typedef __attribute__((ext_vector_type(16))) __bf16 bf16x16;
typedef __attribute__((ext_vector_type(8)))  __bf16 bf16x8;
typedef __attribute__((ext_vector_type(2)))  __bf16 bf16x2;
typedef __attribute__((ext_vector_type(8)))  float  f32x8;
typedef __attribute__((ext_vector_type(4)))  float  f32x4;

static __device__ __forceinline__ f32x8 wmma_bf16(bf16x16 a, bf16x16 b, f32x8 c) {
  // 8 args: (neg_a, A, neg_b, B, c_mod, C, reuse_a, reuse_b)
  return __builtin_amdgcn_wmma_f32_16x16x32_bf16(false, a, false, b, (short)0, c,
                                                 false, false);
}

static __device__ __forceinline__ bf16x16 cat(bf16x8 lo, bf16x8 hi) {
  return __builtin_shufflevector(lo, hi, 0,1,2,3,4,5,6,7,8,9,10,11,12,13,14,15);
}

// Fragment loader: element e of lane (half,idx) <-> k = e%8 + 8*half + 16*(e/8)
// => two contiguous bf16x8 loads at +8*half and +16+8*half.
static __device__ __forceinline__ bf16x16 frag(const __bf16* rowbase, int half) {
  const __bf16* p = rowbase + 8 * half;
  return cat(*(const bf16x8*)p, *(const bf16x8*)(p + 16));
}

// ---------------------------------------------------------------------------
// Prep A: f32 -> bf16 weight conversion (row-major preserved)
// ---------------------------------------------------------------------------
__global__ __launch_bounds__(256) void wcvt_kernel(const float* __restrict__ src,
                                                   __bf16* __restrict__ dst,
                                                   int nelem) {
  int i = (blockIdx.x * 256 + threadIdx.x) * 8;
  if (i >= nelem) return;
  f32x4 a = *(const f32x4*)(src + i);
  f32x4 b = *(const f32x4*)(src + i + 4);
  bf16x8 r;
  r[0] = (__bf16)a[0]; r[1] = (__bf16)a[1]; r[2] = (__bf16)a[2]; r[3] = (__bf16)a[3];
  r[4] = (__bf16)b[0]; r[5] = (__bf16)b[1]; r[6] = (__bf16)b[2]; r[7] = (__bf16)b[3];
  *(bf16x8*)(dst + i) = r;
}

// ---------------------------------------------------------------------------
// Prep B: expand attention bias into S^T tile-register order:
//   biasT[h][nt(13)][mt(14)][lane(32)][r(8)]  (f32)
// lane=(half,idx): n = nt*16 + idx, m = mt*16 + 8*half + r.
// Key mask baked in: m >= 196 -> -1e30; padded query rows (n >= 196) -> 0.
// ---------------------------------------------------------------------------
__global__ __launch_bounds__(256) void bias_expand_kernel(
    const float* __restrict__ ab, const int* __restrict__ bidx, int n_off,
    float* __restrict__ biasT) {
  int i = blockIdx.x * 256 + threadIdx.x;      // over 8*13*14*32*8 = 372,736
  if (i >= HEADS * NT13 * 14 * 32 * 8) return;
  const int r    = i & 7;
  const int lane = (i >> 3) & 31;
  int rest = i >> 8;
  const int mt = rest % 14;  rest /= 14;
  const int nt = rest % NT13;
  const int h  = rest / NT13;
  const int n = nt * 16 + (lane & 15);
  const int m = mt * 16 + 8 * (lane >> 4) + r;
  float v;
  if (m >= NRES)      v = -1.0e30f;            // masked (padded) key
  else if (n >= NRES) v = 0.0f;                // padded query row
  else                v = ab[(size_t)h * n_off + bidx[n * NRES + m]];
  biasT[i] = v;
}

// ---------------------------------------------------------------------------
// Prep C: transpose+convert x [B][512][196] f32 -> xt [B][224][512] bf16,
// zero-filling n >= 196. LDS tile transpose, branch-free boundary handling.
// grid (7 n-tiles of 32, 8 c-tiles of 64, 128 b), block 256
// ---------------------------------------------------------------------------
__global__ __launch_bounds__(256) void xT_kernel(const float* __restrict__ x,
                                                 __bf16* __restrict__ xt) {
  const int nt = blockIdx.x, ct = blockIdx.y, b = blockIdx.z;
  const int tid = threadIdx.x;
  const int n0 = nt * 32;

  __shared__ __bf16 lds[32 * 72];              // [32 n][64 c], stride 72

  // read phase: c-pair x 4 n per thread -> packed b32 LDS stores
  {
    const int c0   = (tid >> 3) * 2;           // 0,2,..,62
    const int nseg = (tid & 7) * 4;            // 0,4,..,28
    const int s = n0 + nseg;
    const float* row0 = x + ((size_t)b * DIMC + ct * 64 + c0) * NRES;
    const float* row1 = row0 + NRES;
    const int sc = min(s, NRES - 4);           // clamped, aligned, in-bounds
    f32x4 a = *(const f32x4*)(row0 + sc);
    f32x4 c = *(const f32x4*)(row1 + sc);
#pragma unroll
    for (int j = 0; j < 4; ++j) {
      bool ok = (s + j < NRES);
      bf16x2 pk;
      pk[0] = ok ? (__bf16)a[j] : (__bf16)0.0f;
      pk[1] = ok ? (__bf16)c[j] : (__bf16)0.0f;
      *(bf16x2*)&lds[(nseg + j) * 72 + c0] = pk;
    }
  }
  __syncthreads();
  // write phase: 32 rows (n) x 64 cols (c); thread: n_l = tid/8, 8 c each
  {
    const int n_l = tid >> 3;
    const int c8  = (tid & 7) * 8;
    bf16x8 v = *(const bf16x8*)&lds[n_l * 72 + c8];
    __bf16* dst = xt + ((size_t)b * NP + n0 + n_l) * DIMC + ct * 64 + c8;
    *(bf16x8*)dst = v;
  }
}

// ---------------------------------------------------------------------------
// Kernel 1: qkv = wq @ x + qkv_b  (per batch: [1536,512] x [512,224])
// Pure register GEMM (no LDS). Stores Q^T [bh][n][64], K^T [bh][m][64],
// V [bh][64][m] (bf16).
// grid (14 n-tiles, 12 o-blocks(128), 128 b), block 256 (8 waves)
// ---------------------------------------------------------------------------
__global__ __launch_bounds__(256) void qkv_kernel(
    const __bf16* __restrict__ xt,     // [B][NP][512]
    const __bf16* __restrict__ wq,     // [1536][512] bf16
    const float*  __restrict__ qkv_b,  // [1536]
    __bf16* __restrict__ qt,           // [B*H][NP][64]
    __bf16* __restrict__ kt,           // [B*H][NP][64]
    __bf16* __restrict__ vv)           // [B*H][64][NP]
{
  const int nt = blockIdx.x, ob = blockIdx.y, b = blockIdx.z;
  const int wave = threadIdx.x >> 5, lane = threadIdx.x & 31;
  const int half = lane >> 4, idx = lane & 15;
  const int n0 = nt * 16;
  const int o0 = ob * 128 + wave * 16;

  const __bf16* abase = wq + (size_t)(o0 + idx) * DIMC;
  const __bf16* bbase = xt + ((size_t)b * NP + n0 + idx) * DIMC;

  f32x8 acc = {};
#pragma unroll 4
  for (int ks = 0; ks < 16; ++ks) {
    const int k0 = ks * 32;
    acc = wmma_bf16(frag(abase + k0, half), frag(bbase + k0, half), acc);
  }

  // epilogue: rows o = o0 + 8*half + r, col n = n0 + idx
  const int sec = o0 >> 9;             // 0=q 1=k 2=v
  const int oc  = o0 & 511;
  const int h   = oc >> 6;
  const int d0  = (oc & 63) + 8 * half;
  const int bh  = b * HEADS + h;

  float bias[8];
#pragma unroll
  for (int r = 0; r < 8; ++r) bias[r] = qkv_b[o0 + 8 * half + r];

  if (sec < 2) {
    __bf16* plane = (sec == 0) ? qt : kt;
    __bf16* dst = plane + ((size_t)bh * NP + (n0 + idx)) * HD + d0;
    bf16x8 pk;
#pragma unroll
    for (int r = 0; r < 8; ++r) pk[r] = (__bf16)(acc[r] + bias[r]);
    *(bf16x8*)dst = pk;                 // transposed 16B store
  } else {
    __bf16* dst = vv + ((size_t)bh * HD + d0) * NP + (n0 + idx);
#pragma unroll
    for (int r = 0; r < 8; ++r) dst[(size_t)r * NP] = (__bf16)(acc[r] + bias[r]);
  }
}

// ---------------------------------------------------------------------------
// Kernel 2: per (bh, 16-row n-tile):
//   S^T = K^T Q (so lane-cols = n), *scale + biasT, in-lane softmax,
//   P -> LDS [n][m] via packed b128 stores, O = V P^T -> Out^T [b][n][512]
// grid (13 n-tiles, 1024 bh), block 32 (one wave)
// ---------------------------------------------------------------------------
__global__ __launch_bounds__(32) void attn_kernel(
    const __bf16* __restrict__ qt,     // [bh][NP][64]
    const __bf16* __restrict__ kt,     // [bh][NP][64]
    const __bf16* __restrict__ vv,     // [bh][64][NP]
    const float*  __restrict__ biasT,  // [H][13][14][32][8], mask baked in
    __bf16* __restrict__ outT)         // [B][NP][512]
{
  const int ntile = blockIdx.x;
  const int bh    = blockIdx.y;
  const int b = bh >> 3, h = bh & 7;
  const int lane = threadIdx.x;
  const int half = lane >> 4, idx = lane & 15;
  const int n0 = ntile * 16;

  __shared__ __bf16 ldsP[16 * 232];    // P tile [16 n][224 m], stride 232

  // Q fragments = B operand of S^T (B^T = Q^T [n][d] row-major)
  const __bf16* qbase = qt + ((size_t)bh * NP + n0 + idx) * HD;
  bf16x16 qf0 = frag(qbase,      half);
  bf16x16 qf1 = frag(qbase + 32, half);

  const float scale = 0.125f;          // 64^-0.5
  const float* bbase =
      biasT + (((size_t)(h * NT13 + ntile) * 14) * 32 + lane) * 8;

  // S^T tiles: rows m = mt*16 + 8*half + r, col n = n0 + idx (lane-owned row)
  f32x8 s[14];
#pragma unroll
  for (int mt = 0; mt < 14; ++mt) {
    const __bf16* kbase = kt + ((size_t)bh * NP + mt * 16 + idx) * HD;
    f32x8 a = {};
    a = wmma_bf16(frag(kbase,      half), qf0, a);
    a = wmma_bf16(frag(kbase + 32, half), qf1, a);
    f32x8 bv = *(const f32x8*)(bbase + (size_t)mt * 256);
#pragma unroll
    for (int r = 0; r < 8; ++r)
      a[r] = fmaf(a[r], scale, bv[r]);
    s[mt] = a;
  }

  // softmax: each lane owns half of row n; other half in lane idx^16
  float mx = -3.0e38f;
#pragma unroll
  for (int mt = 0; mt < 14; ++mt)
#pragma unroll
    for (int r = 0; r < 8; ++r) mx = fmaxf(mx, s[mt][r]);
  mx = fmaxf(mx, __shfl_xor(mx, 16, 32));
  float sum = 0.0f;
#pragma unroll
  for (int mt = 0; mt < 14; ++mt)
#pragma unroll
    for (int r = 0; r < 8; ++r) {
      float e = __expf(s[mt][r] - mx);
      s[mt][r] = e;
      sum += e;
    }
  sum += __shfl_xor(sum, 16, 32);
  const float inv = 1.0f / sum;

  // P -> LDS [n][m]: lane's 8 regs are 8 consecutive m -> one b128 per tile
#pragma unroll
  for (int mt = 0; mt < 14; ++mt) {
    bf16x8 pk;
#pragma unroll
    for (int r = 0; r < 8; ++r) pk[r] = (__bf16)(s[mt][r] * inv);
    *(bf16x8*)&ldsP[idx * 232 + mt * 16 + 8 * half] = pk;
  }
  __syncthreads();

  // O = V * P^T : A = V [d][m] (global, row-major), B^T = P [n][m] (LDS)
  const __bf16* vplane = vv + (size_t)bh * HD * NP;
  const int n = n0 + idx;
#pragma unroll
  for (int dt = 0; dt < 4; ++dt) {
    f32x8 acc = {};
#pragma unroll
    for (int mt = 0; mt < 7; ++mt) {
      const int m0 = mt * 32;
      acc = wmma_bf16(frag(vplane + (size_t)(dt * 16 + idx) * NP + m0, half),
                      frag(&ldsP[idx * 232 + m0], half), acc);
    }
    // store Out^T[b][n][h*64 + d], zero-fill padded rows (read by proj)
    __bf16* dst = outT + ((size_t)b * NP + n) * DIMC + h * HD + dt * 16 + 8 * half;
    bf16x8 pk;
#pragma unroll
    for (int r = 0; r < 8; ++r)
      pk[r] = (n < NRES) ? (__bf16)acc[r] : (__bf16)0.0f;
    *(bf16x8*)dst = pk;
  }
}

// ---------------------------------------------------------------------------
// Kernel 3: y = wp @ Out + proj_b   -> d_out f32 [B][512][196]
// grid (13 n-tiles, 4 o-blocks(128), 128 b), block 256 (8 waves), no LDS
// ---------------------------------------------------------------------------
__global__ __launch_bounds__(256) void proj_kernel(
    const __bf16* __restrict__ outT,   // [B][NP][512]
    const __bf16* __restrict__ wp,     // [512][512] bf16
    const float*  __restrict__ proj_b, // [512]
    float* __restrict__ y)             // [B,512,196]
{
  const int nt = blockIdx.x, b = blockIdx.z;
  const int wave = threadIdx.x >> 5, lane = threadIdx.x & 31;
  const int half = lane >> 4, idx = lane & 15;
  const int o0 = blockIdx.y * 128 + wave * 16;
  const int n0 = nt * 16;

  const __bf16* abase = wp + (size_t)(o0 + idx) * DIMC;
  const __bf16* bbase = outT + ((size_t)b * NP + n0 + idx) * DIMC;

  f32x8 acc = {};
#pragma unroll 4
  for (int ks = 0; ks < 16; ++ks) {
    const int k0 = ks * 32;
    acc = wmma_bf16(frag(abase + k0, half), frag(bbase + k0, half), acc);
  }

  const int n = n0 + idx;
  if (n < NRES) {
#pragma unroll
    for (int r = 0; r < 8; ++r) {
      const int o = o0 + 8 * half + r;
      y[((size_t)b * DIMC + o) * NRES + n] = acc[r] + proj_b[o];
    }
  }
}

// ---------------------------------------------------------------------------
extern "C" void kernel_launch(void* const* d_in, const int* in_sizes, int n_in,
                              void* d_out, int out_size, void* d_ws, size_t ws_size,
                              hipStream_t stream) {
  const float* x      = (const float*)d_in[0];
  const float* qkv_w  = (const float*)d_in[1];
  const float* qkv_b  = (const float*)d_in[2];
  const float* proj_w = (const float*)d_in[3];
  const float* proj_b = (const float*)d_in[4];
  const float* ab     = (const float*)d_in[5];
  const int*   bidx   = (const int*)d_in[6];
  const int n_off = in_sizes[5] / HEADS;

  // workspace carve (total ~143.4 MB)
  const size_t plane = (size_t)BATCH * HEADS * NP * HD;   // 14,680,064 elems
  __bf16* xt   = (__bf16*)d_ws;                 // [B][NP][512]
  __bf16* qt   = xt + plane;
  __bf16* kt   = qt + plane;
  __bf16* vv   = kt + plane;
  __bf16* outT = vv + plane;                    // [B][NP][512]
  __bf16* wq   = outT + plane;                  // 1536*512
  __bf16* wp   = wq + (size_t)OQKV * DIMC;      // 512*512
  float*  biasT = (float*)(wp + (size_t)DIMC * DIMC);  // [8][13][14][32][8]
  (void)ws_size; (void)n_in; (void)out_size;

  wcvt_kernel<<<dim3((OQKV * DIMC) / 2048), 256, 0, stream>>>(qkv_w, wq,
                                                              OQKV * DIMC);
  wcvt_kernel<<<dim3((DIMC * DIMC) / 2048), 256, 0, stream>>>(proj_w, wp,
                                                              DIMC * DIMC);
  const int nbias = HEADS * NT13 * 14 * 32 * 8;
  bias_expand_kernel<<<dim3((nbias + 255) / 256), 256, 0, stream>>>(
      ab, bidx, n_off, biasT);
  xT_kernel<<<dim3(7, 8, BATCH), 256, 0, stream>>>(x, xt);

  qkv_kernel <<<dim3(14, 12, BATCH), 256, 0, stream>>>(xt, wq, qkv_b, qt, kt, vv);
  attn_kernel<<<dim3(13, BATCH * HEADS), 32, 0, stream>>>(qt, kt, vv, biasT, outT);
  proj_kernel<<<dim3(13, 4, BATCH), 256, 0, stream>>>(outT, wp, proj_b,
                                                      (float*)d_out);
}
